// SpikingCNNRegressor_86071144611853
// MI455X (gfx1250) — matrix operand order repaired and checked by reading, not compiled, not run on hardware
//
#include <hip/hip_runtime.h>
#include <hip/hip_bf16.h>

// ---------------------------------------------------------------------------
// SpikingCNNRegressor for MI455X (gfx1250, wave32, WMMA + async-to-LDS)
//   x[64,1024,8] -> conv0(8->256,k3) -> BN -> LIF
//   -> 3x { conv(256->256,k3) as WMMA f16 GEMM -> BN -> LIF }
//   -> mean over T -> head [256->63]
// Conv biases are dropped (cancelled exactly by batch-stat BatchNorm).
// ---------------------------------------------------------------------------

typedef _Float16 v16h __attribute__((ext_vector_type(16)));
typedef _Float16 h8   __attribute__((ext_vector_type(8)));
typedef float    v8f  __attribute__((ext_vector_type(8)));
typedef int      v4i  __attribute__((ext_vector_type(4)));

#define BATCH 64
#define TLEN  1024
#define CIN   8
#define EDIM  256
#define ODIM  63
#define TPAD  (TLEN + 2)      // zero pad row at both ends per batch
#define MROWS (BATCH * TLEN)  // 65536
#define KC    32              // K-chunk staged in LDS per pipeline stage

#if defined(__gfx1250__) &&                                        \
    __has_builtin(__builtin_amdgcn_global_load_async_to_lds_b128) && \
    __has_builtin(__builtin_amdgcn_s_wait_asynccnt)
#define USE_ASYNC 1
#else
#define USE_ASYNC 0
#endif

#define AS1 __attribute__((address_space(1)))
#define AS3 __attribute__((address_space(3)))

// 16-byte global -> LDS copy (async DMA on gfx1250 if available)
__device__ __forceinline__ void cp16(_Float16* dst, const _Float16* src) {
#if USE_ASYNC
  __builtin_amdgcn_global_load_async_to_lds_b128((AS1 v4i*)src,
                                                 (AS3 v4i*)dst, 0, 0);
#else
  *(h8*)dst = *(const h8*)src;
#endif
}

__device__ __forceinline__ void wait_async_done() {
#if USE_ASYNC
  __builtin_amdgcn_s_wait_asynccnt(0);
#endif
}

// ---------------------------------------------------------------------------
// Repack convs_w [3][E][E][3] (layer, e_out, c_in, k) fp32
//   -> wt [3 layers][3 k][e_out][c_in] f16   (B-matrix friendly: N-major)
// ---------------------------------------------------------------------------
__global__ __launch_bounds__(256) void repack_w_kernel(
    const float* __restrict__ cw, _Float16* __restrict__ wt) {
  int i = blockIdx.x * 256 + threadIdx.x;   // enumerates [L][k][e][c]
  if (i >= 3 * 3 * EDIM * EDIM) return;
  int c = i & 255;
  int e = (i >> 8) & 255;
  int rem = i >> 16;          // L*3 + k
  int k = rem % 3;
  int L = rem / 3;
  wt[i] = (_Float16)cw[(((size_t)L * EDIM + e) * EDIM + c) * 3 + k];
}

// ---------------------------------------------------------------------------
// Block 0 conv: y[m, e] = sum_{k,c} x[b, t+k-1, c] * w0[e, c, k]   (fp32)
// one block per m = b*T + t, one thread per e
// ---------------------------------------------------------------------------
__global__ __launch_bounds__(256) void conv0_kernel(
    const float* __restrict__ x, const float* __restrict__ w,
    float* __restrict__ y) {
  int m = blockIdx.x;
  int e = threadIdx.x;
  int b = m >> 10;
  int t = m & 1023;
  float acc = 0.f;
#pragma unroll
  for (int k = 0; k < 3; ++k) {
    int tt = t + k - 1;
    if (tt < 0 || tt >= TLEN) continue;
    const float* xr = x + ((size_t)b * TLEN + tt) * CIN;
    const float* wr = w + (size_t)e * CIN * 3 + k;   // w[e][c][k], c-stride 3
#pragma unroll
    for (int c = 0; c < CIN; ++c) acc += xr[c] * wr[c * 3];
  }
  y[(size_t)m * EDIM + e] = acc;
}

// ---------------------------------------------------------------------------
// BN statistics: coalesced slab reduction, thread <-> channel
// ---------------------------------------------------------------------------
__global__ __launch_bounds__(512) void zero_stats_kernel(float* __restrict__ p) {
  p[threadIdx.x] = 0.f;   // 512 floats: sums[256] ++ sqs[256]
}

__global__ __launch_bounds__(256) void bn_stats_kernel(
    const float* __restrict__ y, float* __restrict__ sums,
    float* __restrict__ sqs) {
  int e = threadIdx.x;
  int m0 = blockIdx.x * 256;
  float s = 0.f, q = 0.f;
  const float* p = y + (size_t)m0 * EDIM + e;
  for (int r = 0; r < 256; ++r) {
    float v = p[(size_t)r * EDIM];
    s += v;
    q += v * v;
  }
  atomicAdd(&sums[e], s);
  atomicAdd(&sqs[e], q);
}

__global__ __launch_bounds__(256) void bn_finalize_kernel(
    const float* __restrict__ sums, const float* __restrict__ sqs,
    const float* __restrict__ gamma, const float* __restrict__ beta,
    float* __restrict__ scale, float* __restrict__ shift) {
  int e = threadIdx.x;
  const float inv_n = 1.f / (float)MROWS;
  float mean = sums[e] * inv_n;
  float var = sqs[e] * inv_n - mean * mean;
  float sc = gamma[e] * rsqrtf(var + 1e-5f);
  scale[e] = sc;
  shift[e] = beta[e] - mean * sc;
}

// ---------------------------------------------------------------------------
// LIF: v' = v + (y_bn - v)/2 ; s = (v' >= 1) ; v = s ? 0 : v'
// Writes spikes f16 into padded [B][T+2][E] layout (zero rows at t=0, T+1).
// Optionally fuses the final temporal mean (pooled != nullptr).
// ---------------------------------------------------------------------------
__global__ __launch_bounds__(256) void lif_kernel(
    const float* __restrict__ y, const float* __restrict__ scale,
    const float* __restrict__ shift, _Float16* __restrict__ spk,
    float* __restrict__ pooled) {
  int e = threadIdx.x;
  int b = blockIdx.x;
  float sc = scale[e], sh = shift[e];
  spk[((size_t)b * TPAD + 0) * EDIM + e] = (_Float16)0.f;
  spk[((size_t)b * TPAD + (TPAD - 1)) * EDIM + e] = (_Float16)0.f;
  const float* yb = y + (size_t)b * TLEN * EDIM + e;
  _Float16* sb = spk + ((size_t)b * TPAD + 1) * EDIM + e;
  float v = 0.f, cnt = 0.f;
  for (int t = 0; t < TLEN; ++t) {
    float yn = yb[(size_t)t * EDIM] * sc + sh;
    v = v + (yn - v) * 0.5f;
    float s = (v >= 1.f) ? 1.f : 0.f;
    v = (s > 0.f) ? 0.f : v;
    cnt += s;
    sb[(size_t)t * EDIM] = (_Float16)s;
  }
  if (pooled) pooled[(size_t)b * EDIM + e] = cnt * (1.f / (float)TLEN);
}

// ---------------------------------------------------------------------------
// Conv (E->E, k=3) as 3 shifted WMMA GEMMs with double-buffered async
// global->LDS staging.
//   A: spikes padded [B][T+2][E] f16; shift k offsets the row base by k*E.
//   B: wt [k][e_out][c] f16 (N-major, so per-lane K runs are contiguous).
//   Y: fp32 [M][E].
// Workgroup: 256 thr = 8 waves; tile 64(M) x 256(N); waves 4(M) x 2(N);
// each wave: 16 x 128 -> 8 accumulator tiles of 16x16 (v8f each).
// Pipeline per 32-wide K-chunk (24 chunks = 3 shifts x 8):
//   s_wait_asynccnt 0 ; barrier ; issue chunk+1 async ; 8x WMMA on chunk.
// ---------------------------------------------------------------------------
__global__ __launch_bounds__(256) void conv_wmma_kernel(
    const _Float16* __restrict__ S,   // padded spikes
    const _Float16* __restrict__ W,   // this layer's wt: [3][256][256]
    float* __restrict__ Y) {
  __shared__ __attribute__((aligned(32))) _Float16 lA[2][64 * KC];   // 2x4 KB
  __shared__ __attribute__((aligned(32))) _Float16 lB[2][EDIM * KC]; // 2x16 KB

  const int tid = threadIdx.x;
  const int lane = tid & 31;
  const int wave = tid >> 5;
  const int g = lane >> 4;    // K-half select
  const int lr = lane & 15;   // row (A) / col (B,D) within tile

  const int mBase = blockIdx.x * 64;  // 16-row tiles never cross a batch (64|1024)
  const int b = mBase >> 10;
  const int t0 = mBase & 1023;
  const int mRow = (wave & 3) * 16;   // wave row offset within 64-row tile
  const int nBase = (wave >> 2) * 128;

  v8f acc[8];
#pragma unroll
  for (int i = 0; i < 8; ++i) acc[i] = (v8f){0.f, 0.f, 0.f, 0.f, 0.f, 0.f, 0.f, 0.f};

  // issue all global->LDS copies for one K-chunk (5 x 16B per thread)
  auto issue = [&](int chunk, int buf) {
    const int sh = chunk >> 3;          // 8 chunks per shift
    const int c0 = (chunk & 7) * KC;
    const _Float16* Ab = S + ((size_t)b * TPAD + t0 + sh) * EDIM + c0;
    const _Float16* Bb = W + (size_t)sh * EDIM * EDIM + c0;
    {  // A: 64 rows x KC halfs = 256 x 16B -> 1 op/thread
      int r = tid >> 2, cc = (tid & 3) * 8;
      cp16(&lA[buf][r * KC + cc], Ab + (size_t)r * EDIM + cc);
    }
#pragma unroll
    for (int i = 0; i < 4; ++i) {  // B: 256 rows x KC halfs -> 4 ops/thread
      int t2 = tid + i * 256;
      int e = t2 >> 2, cc = (t2 & 3) * 8;
      cp16(&lB[buf][e * KC + cc], Bb + (size_t)e * EDIM + cc);
    }
  };

  const int NCHUNK = 24;
  issue(0, 0);
  for (int chunk = 0; chunk < NCHUNK; ++chunk) {
    const int buf = chunk & 1;
    wait_async_done();   // this wave's loads for `chunk` have landed in LDS
    __syncthreads();     // -> all waves' loads landed; prev compute done
    if (chunk + 1 < NCHUNK) issue(chunk + 1, buf ^ 1);  // overlap with compute

    // A fragment (16x32 f16): lane-half g holds K = g*16+0..15
    const int arow = (mRow + lr) * KC;
    h8 a0 = *(const h8*)&lA[buf][arow + g * 16];
    h8 a1 = *(const h8*)&lA[buf][arow + g * 16 + 8];
    v16h afrag = __builtin_shufflevector(a0, a1, 0, 1, 2, 3, 4, 5, 6, 7,
                                         8, 9, 10, 11, 12, 13, 14, 15);
#pragma unroll
    for (int nt = 0; nt < 8; ++nt) {
      // B fragment (32x16 f16): lane = col n, 16 contiguous K per lane-half
      int n = nBase + nt * 16 + lr;
      v16h bfrag = *(const v16h*)&lB[buf][n * KC + g * 16];
      acc[nt] = __builtin_amdgcn_wmma_f32_16x16x32_f16(
          false, afrag, false, bfrag, (short)0, acc[nt], false, false);
    }
  }

  // ---- epilogue: D layout lane=N col, VGPR r -> M row (g*8 + r) ----
#pragma unroll
  for (int nt = 0; nt < 8; ++nt) {
    int n = nBase + nt * 16 + lr;
#pragma unroll
    for (int r = 0; r < 8; ++r) {
      int m = mBase + mRow + g * 8 + r;
      Y[(size_t)m * EDIM + n] = acc[nt][r];
    }
  }
}

// ---------------------------------------------------------------------------
// Head: out[b,o] = pooled[b,:] . head_w[o,:] + head_b[o]
// ---------------------------------------------------------------------------
__global__ __launch_bounds__(256) void head_kernel(
    const float* __restrict__ pooled, const float* __restrict__ hw,
    const float* __restrict__ hb, float* __restrict__ out) {
  int i = blockIdx.x * 256 + threadIdx.x;
  if (i >= BATCH * ODIM) return;
  int b = i / ODIM, o = i % ODIM;
  const float* p = pooled + (size_t)b * EDIM;
  const float* w = hw + (size_t)o * EDIM;
  float a = 0.f;
  for (int e = 0; e < EDIM; ++e) a += p[e] * w[e];
  out[i] = a + hb[o];
}

// ---------------------------------------------------------------------------
extern "C" void kernel_launch(void* const* d_in, const int* in_sizes, int n_in,
                              void* d_out, int out_size, void* d_ws,
                              size_t ws_size, hipStream_t stream) {
  (void)in_sizes; (void)n_in; (void)out_size; (void)ws_size;
  const float* x        = (const float*)d_in[0];
  const float* conv0_w  = (const float*)d_in[1];
  // d_in[2] conv0_b, d_in[4] convs_b: cancelled exactly by batch-stat BN
  const float* convs_w  = (const float*)d_in[3];
  const float* bn_gamma = (const float*)d_in[5];
  const float* bn_beta  = (const float*)d_in[6];
  const float* head_w   = (const float*)d_in[7];
  const float* head_b   = (const float*)d_in[8];
  float* out = (float*)d_out;

  // workspace layout
  char* ws = (char*)d_ws;
  size_t off = 0;
  float* y = (float*)(ws + off);                 off += (size_t)MROWS * EDIM * 4;        // 64 MB
  _Float16* spikes = (_Float16*)(ws + off);      off += (size_t)BATCH * TPAD * EDIM * 2; // 32 MB
  _Float16* wt = (_Float16*)(ws + off);          off += (size_t)3 * 3 * EDIM * EDIM * 2; // 1.1 MB
  float* sums  = (float*)(ws + off);             off += EDIM * 4;
  float* sqs   = (float*)(ws + off);             off += EDIM * 4;
  float* scale = (float*)(ws + off);             off += EDIM * 4;
  float* shift = (float*)(ws + off);             off += EDIM * 4;
  float* pooled = (float*)(ws + off);            off += (size_t)BATCH * EDIM * 4;

  repack_w_kernel<<<(3 * 3 * EDIM * EDIM + 255) / 256, 256, 0, stream>>>(convs_w, wt);
  conv0_kernel<<<MROWS, 256, 0, stream>>>(x, conv0_w, y);

  for (int L = 0; L < 4; ++L) {
    zero_stats_kernel<<<1, 512, 0, stream>>>(sums);  // sums ++ sqs contiguous
    bn_stats_kernel<<<MROWS / 256, 256, 0, stream>>>(y, sums, sqs);
    bn_finalize_kernel<<<1, 256, 0, stream>>>(sums, sqs, bn_gamma + L * EDIM,
                                              bn_beta + L * EDIM, scale, shift);
    lif_kernel<<<BATCH, 256, 0, stream>>>(y, scale, shift, spikes,
                                          (L == 3) ? pooled : nullptr);
    if (L < 3) {
      conv_wmma_kernel<<<MROWS / 64, 256, 0, stream>>>(
          spikes, wt + (size_t)L * 3 * EDIM * EDIM, y);
    }
  }
  head_kernel<<<(BATCH * ODIM + 255) / 256, 256, 0, stream>>>(pooled, head_w,
                                                              head_b, out);
}
